// SAGE_7687991460411
// MI455X (gfx1250) — matrix operand order repaired and checked by reading, not compiled, not run on hardware
//
#include <hip/hip_runtime.h>
#include <hip/hip_bf16.h>
#include <math.h>

// ---------------------------------------------------------------------------
// GraphSAGE 3-layer forward for MI455X (gfx1250, wave32, WMMA bf16->f32).
//   per layer:  agg = segment_mean(x[src] -> dst);  y = agg@Wl + b + x@Wr
//   layers 0,1: ReLU.  layer 2: log_softmax over 64 cols.
// Scatter/gather is the roofline bottleneck (~2GB total @ 23.3TB/s); GEMMs
// are fused two-matmuls-per-tile using v_wmma_f32_16x16x32_bf16, with the
// weights pre-converted to bf16 and pre-swizzled into WMMA B-fragment lane
// order so each fragment is one contiguous 32B load per lane.
// ---------------------------------------------------------------------------

#define NN 100000
#define EE 600000
#define DHID 128

typedef __attribute__((ext_vector_type(16))) __bf16 v16bf;
typedef __attribute__((ext_vector_type(8)))  float  v8f;

// ---------------- utility kernels -----------------------------------------

__global__ void fill_zero_f4(float4* __restrict__ p, int n4) {
    int i = blockIdx.x * blockDim.x + threadIdx.x;
    if (i < n4) p[i] = make_float4(0.f, 0.f, 0.f, 0.f);
}

__global__ void degree_kernel(const long long* __restrict__ dst,
                              float* __restrict__ deg, int E) {
    int e = blockIdx.x * blockDim.x + threadIdx.x;
    if (e < E) atomicAdd(&deg[dst[e]], 1.0f);
}

__global__ void invdeg_kernel(float* __restrict__ deg, int n) {
    int i = blockIdx.x * blockDim.x + threadIdx.x;
    if (i < n) deg[i] = 1.0f / fmaxf(deg[i], 1.0f);
}

// one work-item = one (edge, 4-channel chunk): float4 gather + 4 f32 atomics
__global__ void scatter_kernel(const long long* __restrict__ src,
                               const long long* __restrict__ dst,
                               const float* __restrict__ X,
                               float* __restrict__ agg, int E) {
    long long idx = (long long)blockIdx.x * blockDim.x + threadIdx.x;
    if (idx >= (long long)E * 32) return;
    int e = (int)(idx >> 5);
    int c = (int)(idx & 31) << 2;
    long long s = src[e], d = dst[e];
    float4 v = *(const float4*)(X + s * (long long)DHID + c);
    float* a = agg + d * (long long)DHID + c;
    atomicAdd(a + 0, v.x);
    atomicAdd(a + 1, v.y);
    atomicAdd(a + 2, v.z);
    atomicAdd(a + 3, v.w);
}

// ---------------- weight pre-swizzle (f32 row-major -> bf16 B-fragments) ---
// Fragment order: [ktile][tn][lane][e], 16 bf16 (32B) contiguous per lane.
// Lane l supplies col = tn*16 + (l&15), K rows = ktile*32 + (l>=16?8:0) + (e<8 ? e : e+8).
template <int DOUT>
__global__ void prep_wfrag(const float* __restrict__ W, __bf16* __restrict__ Ws) {
    constexpr int NT  = DOUT / 16;
    constexpr int TOT = (DHID / 32) * NT * 32 * 16;   // == DHID*DOUT
    int i = blockIdx.x * blockDim.x + threadIdx.x;
    if (i >= TOT) return;
    int e    = i & 15;
    int lane = (i >> 4) & 31;
    int tn   = (i >> 9) % NT;
    int kt   = (i >> 9) / NT;
    int krow = kt * 32 + ((lane >= 16) ? 8 : 0) + (e < 8 ? e : e + 8);
    int col  = tn * 16 + (lane & 15);
    Ws[i] = (__bf16)W[(long long)krow * DOUT + col];
}

// ---------------- WMMA A-fragment builder ----------------------------------
// A (16x32 bf16, MxK): lane row = lane&15; lo lanes hold K = ka..ka+7 and
// ka+16..ka+23 with ka = k0 (+8 for hi lanes).  Two 32B-aligned float4 pairs.
__device__ __forceinline__ v16bf load_a_frag(const float* __restrict__ rowp,
                                             int ka, float s) {
    float4 f0 = *(const float4*)(rowp + ka);
    float4 f1 = *(const float4*)(rowp + ka + 4);
    float4 f2 = *(const float4*)(rowp + ka + 16);
    float4 f3 = *(const float4*)(rowp + ka + 20);
    v16bf A;
    A[0]  = (__bf16)(f0.x * s); A[1]  = (__bf16)(f0.y * s);
    A[2]  = (__bf16)(f0.z * s); A[3]  = (__bf16)(f0.w * s);
    A[4]  = (__bf16)(f1.x * s); A[5]  = (__bf16)(f1.y * s);
    A[6]  = (__bf16)(f1.z * s); A[7]  = (__bf16)(f1.w * s);
    A[8]  = (__bf16)(f2.x * s); A[9]  = (__bf16)(f2.y * s);
    A[10] = (__bf16)(f2.z * s); A[11] = (__bf16)(f2.w * s);
    A[12] = (__bf16)(f3.x * s); A[13] = (__bf16)(f3.y * s);
    A[14] = (__bf16)(f3.z * s); A[15] = (__bf16)(f3.w * s);
    return A;
}

// ---------------- fused SAGE GEMM: Y = (agg*inv)@Wl + x@Wr + b -------------
template <int DOUT, bool RELU>
__global__ __launch_bounds__(128)
void sage_gemm(const float* __restrict__ agg, const float* __restrict__ inv,
               const v16bf* __restrict__ WlF, const float* __restrict__ xin,
               const v16bf* __restrict__ WrF, const float* __restrict__ bias,
               float* __restrict__ out, int ntiles) {
    constexpr int NT = DOUT / 16;
    int wave = blockIdx.x * (blockDim.x >> 5) + (threadIdx.x >> 5);
    if (wave >= ntiles) return;                 // wave-uniform: EXEC stays all-1
    int tm = wave / NT, tn = wave % NT;
    int lane = threadIdx.x & 31;
    int lrow = lane & 15;
    bool hi  = lane >= 16;
    int row  = tm * 16 + lrow;
    int col  = tn * 16 + lrow;
    float s  = inv[row];

    const float* aRow = agg + (long long)row * DHID;
    const float* xRow = xin + (long long)row * DHID;

    v8f c = {};
#pragma unroll
    for (int kt = 0; kt < DHID / 32; ++kt) {
        int ka = kt * 32 + (hi ? 8 : 0);
        int fb = (kt * NT + tn) * 32 + lane;    // fragment index for this lane
        v16bf A1 = load_a_frag(aRow, ka, s);
        v16bf B1 = WlF[fb];                     // 32B contiguous -> 2x b128
        c = __builtin_amdgcn_wmma_f32_16x16x32_bf16(false, A1, false, B1,
                                                    (short)0, c, false, false);
        v16bf A2 = load_a_frag(xRow, ka, 1.0f);
        v16bf B2 = WrF[fb];
        c = __builtin_amdgcn_wmma_f32_16x16x32_bf16(false, A2, false, B2,
                                                    (short)0, c, false, false);
    }

    float bv = bias[col];
    int mo = hi ? 8 : 0;
#pragma unroll
    for (int r = 0; r < 8; ++r) {
        float v = c[r] + bv;
        if (RELU) v = fmaxf(v, 0.0f);
        out[(long long)(tm * 16 + r + mo) * DOUT + col] = v;
    }
}

// ---------------- row-wise log_softmax over 64 cols ------------------------
__global__ void logsoftmax64(float* __restrict__ y, int n) {
    int r = blockIdx.x * blockDim.x + threadIdx.x;
    if (r >= n) return;
    float* p = y + (long long)r * 64;
    float m = -INFINITY;
    for (int j = 0; j < 64; ++j) m = fmaxf(m, p[j]);
    float ssum = 0.f;
    for (int j = 0; j < 64; ++j) ssum += expf(p[j] - m);
    float l = logf(ssum) + m;
    for (int j = 0; j < 64; ++j) p[j] = p[j] - l;
}

// ---------------------------------------------------------------------------
static inline size_t align256(size_t x) { return (x + 255) & ~(size_t)255; }

extern "C" void kernel_launch(void* const* d_in, const int* in_sizes, int n_in,
                              void* d_out, int out_size, void* d_ws, size_t ws_size,
                              hipStream_t stream) {
    (void)in_sizes; (void)n_in; (void)out_size; (void)ws_size;
    const float*     x    = (const float*)d_in[0];
    const long long* ei   = (const long long*)d_in[1];   // int64 [2, E]
    const long long* src  = ei;
    const long long* dst  = ei + EE;
    const float* Wl0 = (const float*)d_in[2];
    const float* b0  = (const float*)d_in[3];
    const float* Wr0 = (const float*)d_in[4];
    const float* Wl1 = (const float*)d_in[5];
    const float* b1  = (const float*)d_in[6];
    const float* Wr1 = (const float*)d_in[7];
    const float* Wl2 = (const float*)d_in[8];
    const float* b2  = (const float*)d_in[9];
    const float* Wr2 = (const float*)d_in[10];
    float* out = (float*)d_out;

    char* ws = (char*)d_ws;
    size_t o = 0;
    float* deg = (float*)(ws + o); o += align256((size_t)NN * 4);
    float* agg = (float*)(ws + o); o += align256((size_t)NN * DHID * 4);
    float* h0  = (float*)(ws + o); o += align256((size_t)NN * DHID * 4);
    float* h1  = (float*)(ws + o); o += align256((size_t)NN * DHID * 4);
    __bf16* wl0s = (__bf16*)(ws + o); o += align256((size_t)DHID * DHID * 2);
    __bf16* wr0s = (__bf16*)(ws + o); o += align256((size_t)DHID * DHID * 2);
    __bf16* wl1s = (__bf16*)(ws + o); o += align256((size_t)DHID * DHID * 2);
    __bf16* wr1s = (__bf16*)(ws + o); o += align256((size_t)DHID * DHID * 2);
    __bf16* wl2s = (__bf16*)(ws + o); o += align256((size_t)DHID * 64 * 2);
    __bf16* wr2s = (__bf16*)(ws + o);

    const int feat4 = NN * DHID / 4;                   // float4 count of N x 128
    const int deg4  = NN / 4;
    const long long scatterWork = (long long)EE * 32;
    const int scatterBlocks = (int)((scatterWork + 255) / 256);
    const int tiles128 = (NN / 16) * (DHID / 16);      // 50000
    const int tiles64  = (NN / 16) * (64 / 16);        // 25000
    const int w128Blocks = (DHID * DHID + 255) / 256;  // 64
    const int w64Blocks  = (DHID * 64 + 255) / 256;    // 32

    // one-time weight swizzle into WMMA B-fragment order (bf16)
    prep_wfrag<128><<<w128Blocks, 256, 0, stream>>>(Wl0, wl0s);
    prep_wfrag<128><<<w128Blocks, 256, 0, stream>>>(Wr0, wr0s);
    prep_wfrag<128><<<w128Blocks, 256, 0, stream>>>(Wl1, wl1s);
    prep_wfrag<128><<<w128Blocks, 256, 0, stream>>>(Wr1, wr1s);
    prep_wfrag<64><<<w64Blocks, 256, 0, stream>>>(Wl2, wl2s);
    prep_wfrag<64><<<w64Blocks, 256, 0, stream>>>(Wr2, wr2s);

    // degree -> inverse mean scale (computed once, valid all three layers)
    fill_zero_f4<<<(deg4 + 255) / 256, 256, 0, stream>>>((float4*)deg, deg4);
    degree_kernel<<<(EE + 255) / 256, 256, 0, stream>>>(dst, deg, EE);
    invdeg_kernel<<<(NN + 255) / 256, 256, 0, stream>>>(deg, NN);

    // ---- layer 0: x -> h0 (ReLU)
    fill_zero_f4<<<(feat4 + 255) / 256, 256, 0, stream>>>((float4*)agg, feat4);
    scatter_kernel<<<scatterBlocks, 256, 0, stream>>>(src, dst, x, agg, EE);
    sage_gemm<128, true><<<(tiles128 + 3) / 4, 128, 0, stream>>>(
        agg, deg, (const v16bf*)wl0s, x, (const v16bf*)wr0s, b0, h0, tiles128);

    // ---- layer 1: h0 -> h1 (ReLU)
    fill_zero_f4<<<(feat4 + 255) / 256, 256, 0, stream>>>((float4*)agg, feat4);
    scatter_kernel<<<scatterBlocks, 256, 0, stream>>>(src, dst, h0, agg, EE);
    sage_gemm<128, true><<<(tiles128 + 3) / 4, 128, 0, stream>>>(
        agg, deg, (const v16bf*)wl1s, h0, (const v16bf*)wr1s, b1, h1, tiles128);

    // ---- layer 2: h1 -> out (64 cols), then log_softmax in place
    fill_zero_f4<<<(feat4 + 255) / 256, 256, 0, stream>>>((float4*)agg, feat4);
    scatter_kernel<<<scatterBlocks, 256, 0, stream>>>(src, dst, h1, agg, EE);
    sage_gemm<64, false><<<(tiles64 + 3) / 4, 128, 0, stream>>>(
        agg, deg, (const v16bf*)wl2s, h1, (const v16bf*)wr2s, b2, out, tiles64);
    logsoftmax64<<<(NN + 255) / 256, 256, 0, stream>>>(out, NN);
}
